// ContextualTSA_31739808317536
// MI455X (gfx1250) — matrix operand order repaired and checked by reading, not compiled, not run on hardware
//
#include <hip/hip_runtime.h>
#include <hip/hip_bf16.h>
#include <math.h>
#include <stdint.h>

typedef __attribute__((ext_vector_type(16))) _Float16 v16h;
typedef __attribute__((ext_vector_type(8)))  _Float16 v8h;
typedef __attribute__((ext_vector_type(8)))  float    v8f;

#define B_    4
#define L_    2048
#define DM    1024
#define DC    512
#define NLAY  2
#define NS    16
#define DI    1024
#define DTR   32
#define DCONV 4
#define EPSV  1e-5f
#define ROWS  (B_*L_)   // 8192

// ---------------------------------------------------------------------------
// Epilogue helper (templated: straight-line code, no runtime branches)
// ---------------------------------------------------------------------------
template <int ACT, bool HASBIAS, bool HASRES>
__device__ __forceinline__ float epilogue(float v, int row, int col, int N,
                                          const float* __restrict__ bias,
                                          const float* __restrict__ res)
{
    if (HASBIAS) v += bias[col];
    if (HASRES)  v += res[(size_t)row * N + col];
    if (ACT == 1) v = (v > 20.f) ? v : log1pf(__expf(v));
    return v;
}

// ---------------------------------------------------------------------------
// Big WMMA GEMM with double-buffered LDS staging via async-to-LDS (CDNA5).
// C[M,N] = A[M,K] (f16) @ W[N,K]^T (f16) [+bias][+res][softplus]
// Block: 256 threads = 8 waves (4x2). Block tile 128x128; wave tile 32x64
// (2x4 WMMA accumulators). K multiple of 32; M,N multiples of 128.
// ---------------------------------------------------------------------------
template <int ACT, bool HASBIAS, bool HASRES>
__global__ __launch_bounds__(256)
void gemm_big(const _Float16* __restrict__ A, const _Float16* __restrict__ W,
              float* __restrict__ C, int M, int N, int K,
              const float* __restrict__ bias, const float* __restrict__ res)
{
    __shared__ _Float16 As[2][128 * 32];
    __shared__ _Float16 Bs[2][128 * 32];

    const int lane = threadIdx.x & 31;
    const int wave = threadIdx.x >> 5;
    const int wm   = wave >> 1;          // 0..3
    const int wn   = wave & 1;           // 0..1
    const int mblk = blockIdx.y * 128;
    const int nblk = blockIdx.x * 128;
    if (mblk >= M || nblk >= N) return;

    const int lr = lane & 15;
    const int ak = (lane >> 4) * 8;      // A K sub-offset (second lane half +8)
    const int bk = (lane >> 4) * 8;      // B K sub-offset

    v8f zero = {};
    v8f acc[2][4];
    #pragma unroll
    for (int i = 0; i < 2; ++i)
        #pragma unroll
        for (int j = 0; j < 4; ++j) acc[i][j] = zero;

    // cooperative async staging of one 128x32 A tile and one 128x32 B tile
    auto stage = [&](int buf, int k0) {
        #pragma unroll
        for (int t = 0; t < 2; ++t) {
            int chunk = threadIdx.x + t * 256;   // 512 chunks of 16B per tile
            int r  = chunk >> 2;                 // row 0..127
            int co = (chunk & 3) * 8;            // k sub-offset 0/8/16/24
            const _Float16* ga = A + (size_t)(mblk + r) * K + k0 + co;
            uint32_t la = (uint32_t)(uintptr_t)(&As[buf][r * 32 + co]);
            asm volatile("global_load_async_to_lds_b128 %0, %1, off"
                         :: "v"(la), "v"(ga) : "memory");
            const _Float16* gb = W + (size_t)(nblk + r) * K + k0 + co;
            uint32_t lb = (uint32_t)(uintptr_t)(&Bs[buf][r * 32 + co]);
            asm volatile("global_load_async_to_lds_b128 %0, %1, off"
                         :: "v"(lb), "v"(gb) : "memory");
        }
    };

    stage(0, 0);
    asm volatile("s_wait_asynccnt 0" ::: "memory");
    __syncthreads();

    int cur = 0;
    for (int k0 = 0; k0 < K; k0 += 32) {
        if (k0 + 32 < K) stage(cur ^ 1, k0 + 32);

        v16h a[2], b[4];
        #pragma unroll
        for (int i = 0; i < 2; ++i) {
            const _Float16* pa = &As[cur][(wm * 32 + i * 16 + lr) * 32 + ak];
            v8h lo = *(const v8h*)(pa);
            v8h hi = *(const v8h*)(pa + 16);
            #pragma unroll
            for (int e = 0; e < 8; ++e) { a[i][e] = lo[e]; a[i][8 + e] = hi[e]; }
        }
        #pragma unroll
        for (int j = 0; j < 4; ++j) {
            const _Float16* pb = &Bs[cur][(wn * 64 + j * 16 + lr) * 32 + bk];
            v8h lo = *(const v8h*)(pb);
            v8h hi = *(const v8h*)(pb + 16);
            #pragma unroll
            for (int e = 0; e < 8; ++e) { b[j][e] = lo[e]; b[j][8 + e] = hi[e]; }
        }
        #pragma unroll
        for (int i = 0; i < 2; ++i)
            #pragma unroll
            for (int j = 0; j < 4; ++j)
                acc[i][j] = __builtin_amdgcn_wmma_f32_16x16x32_f16(
                    false, a[i], false, b[j], (short)0, acc[i][j], false, false);

        asm volatile("s_wait_asynccnt 0" ::: "memory");
        __syncthreads();
        cur ^= 1;
    }

    const int cn = lane & 15;
    const int cm = (lane >> 4) * 8;
    const int m0 = mblk + wm * 32;
    const int n0 = nblk + wn * 64;
    #pragma unroll
    for (int i = 0; i < 2; ++i)
        #pragma unroll
        for (int j = 0; j < 4; ++j)
            #pragma unroll
            for (int e = 0; e < 8; ++e) {
                int row = m0 + i * 16 + cm + e;
                int col = n0 + j * 16 + cn;
                float v = epilogue<ACT, HASBIAS, HASRES>(acc[i][j][e], row, col, N, bias, res);
                C[(size_t)row * N + col] = v;
            }
}

// ---------------------------------------------------------------------------
// Small WMMA GEMM (direct global fragment loads) for N == 64 (dbc projection).
// Block 128 threads = 4 waves, block tile 64x64, wave tile 32x32.
// ---------------------------------------------------------------------------
template <int ACT, bool HASBIAS, bool HASRES>
__global__ __launch_bounds__(128)
void gemm_small(const _Float16* __restrict__ A, const _Float16* __restrict__ W,
                float* __restrict__ C, int M, int N, int K,
                const float* __restrict__ bias, const float* __restrict__ res)
{
    const int lane = threadIdx.x & 31;
    const int wave = threadIdx.x >> 5;
    const int wm   = wave >> 1;
    const int wn   = wave & 1;
    const int m0   = blockIdx.y * 64 + wm * 32;
    const int n0   = blockIdx.x * 64 + wn * 32;
    if (m0 >= M || n0 >= N) return;

    const int lr = lane & 15;
    const int ak = (lane >> 4) * 8;
    const int bk = (lane >> 4) * 8;

    v8f zero = {};
    v8f acc[2][2];
    acc[0][0] = zero; acc[0][1] = zero; acc[1][0] = zero; acc[1][1] = zero;

    for (int k0 = 0; k0 < K; k0 += 32) {
        v16h a[2], b[2];
        #pragma unroll
        for (int i = 0; i < 2; ++i) {
            const _Float16* pa = A + (size_t)(m0 + i * 16 + lr) * K + k0 + ak;
            v8h alo = *(const v8h*)(pa);
            v8h ahi = *(const v8h*)(pa + 16);
            #pragma unroll
            for (int e = 0; e < 8; ++e) { a[i][e] = alo[e]; a[i][8 + e] = ahi[e]; }
            const _Float16* pb = W + (size_t)(n0 + i * 16 + lr) * K + k0 + bk;
            v8h blo = *(const v8h*)(pb);
            v8h bhi = *(const v8h*)(pb + 16);
            #pragma unroll
            for (int e = 0; e < 8; ++e) { b[i][e] = blo[e]; b[i][8 + e] = bhi[e]; }
        }
        #pragma unroll
        for (int i = 0; i < 2; ++i)
            #pragma unroll
            for (int j = 0; j < 2; ++j)
                acc[i][j] = __builtin_amdgcn_wmma_f32_16x16x32_f16(
                    false, a[i], false, b[j], (short)0, acc[i][j], false, false);
    }

    const int cn = lane & 15;
    const int cm = (lane >> 4) * 8;
    #pragma unroll
    for (int i = 0; i < 2; ++i)
        #pragma unroll
        for (int j = 0; j < 2; ++j)
            #pragma unroll
            for (int e = 0; e < 8; ++e) {
                int row = m0 + i * 16 + cm + e;
                int col = n0 + j * 16 + cn;
                float v = epilogue<ACT, HASBIAS, HASRES>(acc[i][j][e], row, col, N, bias, res);
                C[(size_t)row * N + col] = v;
            }
}

// ---------------------------------------------------------------------------
__global__ void f32_to_f16_k(const float* __restrict__ in, _Float16* __restrict__ out, int n)
{
    int i = blockIdx.x * blockDim.x + threadIdx.x;
    if (i < n) out[i] = (_Float16)in[i];
}

// LayerNorm over DM=1024, one wave per row
__global__ __launch_bounds__(256)
void layernorm_k(const float* __restrict__ src, const float* __restrict__ g,
                 const float* __restrict__ bb, float* __restrict__ s)
{
    int lane = threadIdx.x & 31;
    int row  = blockIdx.x * 8 + (threadIdx.x >> 5);
    const float* x = src + (size_t)row * DM;
    float v[32];
    float sum = 0.f;
    #pragma unroll
    for (int i = 0; i < 32; ++i) { v[i] = x[lane + i * 32]; sum += v[i]; }
    #pragma unroll
    for (int off = 16; off; off >>= 1) sum += __shfl_xor(sum, off, 32);
    float mu = sum * (1.f / DM);
    float vs = 0.f;
    #pragma unroll
    for (int i = 0; i < 32; ++i) { float d = v[i] - mu; vs += d * d; }
    #pragma unroll
    for (int off = 16; off; off >>= 1) vs += __shfl_xor(vs, off, 32);
    float rs = rsqrtf(vs * (1.f / DM) + EPSV);
    float* o = s + (size_t)row * DM;
    #pragma unroll
    for (int i = 0; i < 32; ++i) {
        int e = lane + i * 32;
        o[e] = (v[i] - mu) * rs * g[e] + bb[e];
    }
}

// RMSNorm over DC=512, one wave per row; f16 output
__global__ __launch_bounds__(256)
void rmsnorm_k(const float* __restrict__ h, const float* __restrict__ w,
               _Float16* __restrict__ out)
{
    int lane = threadIdx.x & 31;
    int row  = blockIdx.x * 8 + (threadIdx.x >> 5);
    const float* x = h + (size_t)row * DC;
    float v[16];
    float sum = 0.f;
    #pragma unroll
    for (int i = 0; i < 16; ++i) { v[i] = x[lane + i * 32]; sum += v[i] * v[i]; }
    #pragma unroll
    for (int off = 16; off; off >>= 1) sum += __shfl_xor(sum, off, 32);
    float rs = rsqrtf(sum * (1.f / DC) + EPSV);
    _Float16* o = out + (size_t)row * DC;
    #pragma unroll
    for (int i = 0; i < 16; ++i) {
        int e = lane + i * 32;
        o[e] = (_Float16)(v[i] * rs * w[e]);
    }
}

__global__ void copy_x1_k(const float* __restrict__ s, float* __restrict__ h)
{
    int idx = blockIdx.x * blockDim.x + threadIdx.x;
    if (idx >= ROWS * DC) return;
    int row = idx / DC, c = idx % DC;
    h[idx] = s[(size_t)row * DM + c];
}

// Causal depthwise conv (taps=4) + bias + SiLU
__global__ void conv_silu_k(const float* __restrict__ xz, const float* __restrict__ cw,
                            const float* __restrict__ cb, float* __restrict__ xc,
                            _Float16* __restrict__ xch)
{
    int idx = blockIdx.x * blockDim.x + threadIdx.x;
    if (idx >= ROWS * DI) return;
    int d = idx % DI;
    int row = idx / DI;
    int t = row % L_;
    float acc = cb[d];
    #pragma unroll
    for (int j = 0; j < DCONV; ++j) {
        int tt = t - (DCONV - 1) + j;
        if (tt >= 0)
            acc += cw[d * DCONV + j] * xz[(size_t)(row - (DCONV - 1) + j) * (2 * DI) + d];
    }
    float y = acc / (1.f + __expf(-acc));
    xc[idx]  = y;
    xch[idx] = (_Float16)y;
}

__global__ void dt_extract_k(const float* __restrict__ dbc, _Float16* __restrict__ dth)
{
    int idx = blockIdx.x * blockDim.x + threadIdx.x;
    if (idx >= ROWS * DTR) return;
    int row = idx / DTR, c = idx % DTR;
    dth[idx] = (_Float16)dbc[(size_t)row * 64 + c];
}

// Selective scan: one lane per (b,d) channel, 16 states in registers
__global__ __launch_bounds__(256)
void scan_k(const float* __restrict__ delta, const float* __restrict__ xc,
            const float* __restrict__ dbc, const float* __restrict__ A_log,
            float* __restrict__ y)
{
    int idx = blockIdx.x * blockDim.x + threadIdx.x;   // B_*DI = 4096
    if (idx >= B_ * DI) return;
    int b = idx / DI, d = idx % DI;
    float An[NS];
    #pragma unroll
    for (int n = 0; n < NS; ++n) An[n] = -__expf(A_log[d * NS + n]);
    float hst[NS] = {};
    for (int t = 0; t < L_; ++t) {
        size_t row = (size_t)b * L_ + t;
        float dv = delta[row * DI + d];
        float xv = xc[row * DI + d];
        float dx = dv * xv;
        const float* bc = dbc + row * 64;
        float yv = 0.f;
        #pragma unroll
        for (int n = 0; n < NS; ++n) {
            float dA = __expf(dv * An[n]);
            hst[n] = dA * hst[n] + dx * bc[32 + n];
            yv += hst[n] * bc[48 + n];
        }
        y[row * DI + d] = yv;
    }
}

// y = (scan + Dp*xc) * silu(z) -> f16
__global__ void combine_k(const float* __restrict__ y, const float* __restrict__ xc,
                          const float* __restrict__ xz, const float* __restrict__ Dp,
                          _Float16* __restrict__ yh)
{
    int idx = blockIdx.x * blockDim.x + threadIdx.x;
    if (idx >= ROWS * DI) return;
    int d = idx % DI;
    int row = idx / DI;
    float v = y[idx] + Dp[d] * xc[idx];
    float z = xz[(size_t)row * (2 * DI) + DI + d];
    float sz = z / (1.f + __expf(-z));
    yh[idx] = (_Float16)(v * sz);
}

// out_cat = concat(x1, x2 + h) as f16
__global__ void cat_k(const float* __restrict__ s, const float* __restrict__ h,
                      _Float16* __restrict__ cat)
{
    int idx = blockIdx.x * blockDim.x + threadIdx.x;
    if (idx >= ROWS * DM) return;
    int d = idx % DM;
    int row = idx / DM;
    float v = s[idx];
    if (d >= DC) v += h[(size_t)row * DC + (d - DC)];
    cat[idx] = (_Float16)v;
}

// ---------------------------------------------------------------------------
extern "C" void kernel_launch(void* const* d_in, const int* in_sizes, int n_in,
                              void* d_out, int out_size, void* d_ws, size_t ws_size,
                              hipStream_t stream)
{
    (void)in_sizes; (void)n_in; (void)out_size; (void)ws_size;
    const float* src    = (const float*)d_in[0];
    const float* ln_g   = (const float*)d_in[1];
    const float* ln_b   = (const float*)d_in[2];
    const float* po_W   = (const float*)d_in[3];
    const float* po_b   = (const float*)d_in[4];
    const float* rms_w  = (const float*)d_in[5];
    const float* in_W   = (const float*)d_in[6];
    const float* conv_w = (const float*)d_in[7];
    const float* conv_b = (const float*)d_in[8];
    const float* xp_W   = (const float*)d_in[9];
    const float* dt_W   = (const float*)d_in[10];
    const float* dt_b   = (const float*)d_in[11];
    const float* A_log  = (const float*)d_in[12];
    const float* Dp     = (const float*)d_in[13];
    const float* mo_W   = (const float*)d_in[14];
    float* out = (float*)d_out;

    char* ws = (char*)d_ws;
    size_t off = 0;
    auto alloc = [&](size_t bytes) -> char* {
        char* p = ws + off;
        off += (bytes + 255) & ~(size_t)255;
        return p;
    };
    float*    s    = (float*)   alloc((size_t)ROWS * DM * 4);
    float*    h    = (float*)   alloc((size_t)ROWS * DC * 4);
    float*    xz   = (float*)   alloc((size_t)ROWS * 2 * DI * 4);
    float*    xc   = (float*)   alloc((size_t)ROWS * DI * 4);
    _Float16* xch  = (_Float16*)alloc((size_t)ROWS * DI * 2);
    _Float16* xnh  = (_Float16*)alloc((size_t)ROWS * DC * 2);
    float*    dbc  = (float*)   alloc((size_t)ROWS * 64 * 4);
    _Float16* dth  = (_Float16*)alloc((size_t)ROWS * DTR * 2);
    float*    delta= (float*)   alloc((size_t)ROWS * DI * 4);
    float*    yv   = (float*)   alloc((size_t)ROWS * DI * 4);
    _Float16* yh   = (_Float16*)alloc((size_t)ROWS * DI * 2);
    _Float16* cth  = (_Float16*)alloc((size_t)ROWS * DM * 2);
    _Float16* w_in = (_Float16*)alloc((size_t)NLAY * 2 * DI * DC * 2);
    _Float16* w_xp = (_Float16*)alloc((size_t)NLAY * 64 * DI * 2);
    _Float16* w_dt = (_Float16*)alloc((size_t)NLAY * DI * DTR * 2);
    _Float16* w_mo = (_Float16*)alloc((size_t)NLAY * DC * DI * 2);
    _Float16* w_po = (_Float16*)alloc((size_t)DM * DM * 2);

    auto cvt = [&](const float* a, _Float16* b, int n) {
        f32_to_f16_k<<<(n + 255) / 256, 256, 0, stream>>>(a, b, n);
    };
    cvt(in_W, w_in, NLAY * 2 * DI * DC);
    cvt(xp_W, w_xp, NLAY * 64 * DI);
    cvt(dt_W, w_dt, NLAY * DI * DTR);
    cvt(mo_W, w_mo, NLAY * DC * DI);
    cvt(po_W, w_po, DM * DM);

    layernorm_k<<<ROWS / 8, 256, 0, stream>>>(src, ln_g, ln_b, s);
    copy_x1_k<<<(ROWS * DC + 255) / 256, 256, 0, stream>>>(s, h);

    for (int l = 0; l < NLAY; ++l) {
        rmsnorm_k<<<ROWS / 8, 256, 0, stream>>>(h, rms_w + l * DC, xnh);

        // xz = xn @ in_W^T : M=8192, N=2048, K=512
        {
            dim3 g(2 * DI / 128, ROWS / 128);
            gemm_big<0, false, false><<<g, 256, 0, stream>>>(
                xnh, w_in + (size_t)l * 2 * DI * DC, xz, ROWS, 2 * DI, DC,
                nullptr, nullptr);
        }

        conv_silu_k<<<(ROWS * DI + 255) / 256, 256, 0, stream>>>(
            xz, conv_w + l * DI * DCONV, conv_b + l * DI, xc, xch);

        // dbc = xc @ xp_W^T : M=8192, N=64, K=1024
        {
            dim3 g(1, ROWS / 64);
            gemm_small<0, false, false><<<g, 128, 0, stream>>>(
                xch, w_xp + (size_t)l * 64 * DI, dbc, ROWS, 64, DI,
                nullptr, nullptr);
        }

        dt_extract_k<<<(ROWS * DTR + 255) / 256, 256, 0, stream>>>(dbc, dth);

        // delta = softplus(dt @ dt_W^T + dt_b) : M=8192, N=1024, K=32
        {
            dim3 g(DI / 128, ROWS / 128);
            gemm_big<1, true, false><<<g, 256, 0, stream>>>(
                dth, w_dt + (size_t)l * DI * DTR, delta, ROWS, DI, DTR,
                dt_b + l * DI, nullptr);
        }

        scan_k<<<(B_ * DI + 255) / 256, 256, 0, stream>>>(
            delta, xc, dbc, A_log + (size_t)l * DI * NS, yv);

        combine_k<<<(ROWS * DI + 255) / 256, 256, 0, stream>>>(
            yv, xc, xz, Dp + l * DI, yh);

        // h = h + y @ mo_W^T : M=8192, N=512, K=1024 (residual in-place)
        {
            dim3 g(DC / 128, ROWS / 128);
            gemm_big<0, false, true><<<g, 256, 0, stream>>>(
                yh, w_mo + (size_t)l * DC * DI, h, ROWS, DC, DI,
                nullptr, h);
        }
    }

    cat_k<<<(ROWS * DM + 255) / 256, 256, 0, stream>>>(s, h, cth);

    // out = s + cat @ po_W^T + po_b : M=8192, N=1024, K=1024
    {
        dim3 g(DM / 128, ROWS / 128);
        gemm_big<0, true, true><<<g, 256, 0, stream>>>(
            cth, w_po, out, ROWS, DM, DM, po_b, s);
    }
}
